// Hgnn_17394617548829
// MI455X (gfx1250) — matrix operand order repaired and checked by reading, not compiled, not run on hardware
//
#include <hip/hip_runtime.h>
#include <stdint.h>
#include <stddef.h>

#define N_NODES  50000
#define N_HEDGES 5000
#define NNZ      800000
#define IN_C     128
#define D1       256
#define D2       64

typedef __attribute__((ext_vector_type(16))) __bf16 v16bf;
typedef __attribute__((ext_vector_type(8)))  float  v8f;

__device__ __forceinline__ __bf16 f32_to_bf16(float f) {
    unsigned int u = __builtin_bit_cast(unsigned int, f);
    u += 0x7FFFu + ((u >> 16) & 1u);          // round-to-nearest-even
    unsigned short h = (unsigned short)(u >> 16);
    return __builtin_bit_cast(__bf16, h);
}

// ---------------------------------------------------------------- utility
__global__ __launch_bounds__(256) void k_fill_zero(float* __restrict__ p, int n) {
    int i = blockIdx.x * 256 + threadIdx.x;
    if (i < n) p[i] = 0.0f;
}

__global__ __launch_bounds__(256) void k_degrees(const int* __restrict__ nidx,
                                                 const int* __restrict__ hidx,
                                                 float* __restrict__ deg,
                                                 float* __restrict__ cnt) {
    int i = blockIdx.x * 256 + threadIdx.x;
    if (i < NNZ) {
        atomicAdd(&deg[nidx[i]], 1.0f);
        atomicAdd(&cnt[hidx[i]], 1.0f);
    }
}

__global__ __launch_bounds__(256) void k_cvt_bf16(const float* __restrict__ x,
                                                  __bf16* __restrict__ y, int n) {
    int i = blockIdx.x * 256 + threadIdx.x;
    if (i < n) y[i] = f32_to_bf16(x[i]);
}

// W is KxN row-major f32 -> Wt is NxK row-major bf16 (i.e. column-major W)
__global__ __launch_bounds__(256) void k_cvt_transpose_bf16(const float* __restrict__ W,
                                                            __bf16* __restrict__ Wt,
                                                            int K, int N) {
    int i = blockIdx.x * 256 + threadIdx.x;
    if (i < K * N) {
        int k = i / N, n = i % N;
        Wt[(size_t)n * K + k] = f32_to_bf16(W[i]);
    }
}

// ---------------------------------------------------------------- WMMA GEMM
// C[MxN] = A[MxK] * B[KxN], A row-major bf16, B given as Bt = NxK row-major bf16.
// One wave computes one 16x16 tile of C; K stepped by 32 via v_wmma_f32_16x16x32_bf16.
// ISA fragment layouts (CDNA5 ISA §7.12.2):
//   A 16x32: lanes 0-15 hold M=lane, K = {0..7, 16..23}; lanes 16-31 K = {8..15, 24..31}
//   B 32x16: lanes 0-15 hold N=lane, K = {0..7, 16..23}; lanes 16-31 K = {8..15, 24..31}
//   C/D: VGPR r -> M = (lane>>4)*8 + r, N = lane&15
__global__ __launch_bounds__(256)
void k_gemm_bf16_wmma(const __bf16* __restrict__ A,
                      const __bf16* __restrict__ Bt,
                      float* __restrict__ C,
                      int M, int N, int K) {
    const int lane   = threadIdx.x & 31;
    const int wave   = threadIdx.x >> 5;
    const int tilesN = N >> 4;
    const int tile   = blockIdx.x * 8 + wave;
    if (tile >= (M >> 4) * tilesN) return;     // wave-uniform: EXEC stays all-ones
    const int tm = (tile / tilesN) << 4;
    const int tn = (tile % tilesN) << 4;

    const int m  = tm + (lane & 15);
    const int n  = tn + (lane & 15);
    const int hk = lane >> 4;                  // K half-stripe select

    v8f acc = {};
    for (int k0 = 0; k0 < K; k0 += 32) {
        union { v16bf v; uint4 q[2]; } fa, fb;
        const __bf16* ap = A + (size_t)m * K + k0 + hk * 8;
        fa.q[0] = *(const uint4*)(ap);         // K = k0+hk*8   .. +7
        fa.q[1] = *(const uint4*)(ap + 16);    // K = k0+hk*8+16.. +7
        const __bf16* bp = Bt + (size_t)n * K + k0 + hk * 8;
        fb.q[0] = *(const uint4*)(bp);         // K = k0+hk*8   .. +7
        fb.q[1] = *(const uint4*)(bp + 16);    // K = k0+hk*8+16.. +7
        acc = __builtin_amdgcn_wmma_f32_16x16x32_bf16(
                  false, fa.v, false, fb.v, (short)0, acc, false, false);
    }
    float* cp = C + (size_t)(tm + hk * 8) * N + n;
#pragma unroll
    for (int r = 0; r < 8; ++r) cp[(size_t)r * N] = acc[r];
}

// ---------------------------------------------------------------- sparse aggregation
// dst[didx[e], :] += src[sidx[e], :]   (F channels, multiple of 4)
__global__ __launch_bounds__(256)
void k_scatter_add(const float* __restrict__ src, const int* __restrict__ sidx,
                   float* __restrict__ dst, const int* __restrict__ didx,
                   int nnz, int F) {
    int t = blockIdx.x * 256 + threadIdx.x;
    int chunks = F >> 2;
    if (t >= nnz * chunks) return;
    int e = t / chunks;
    int c = (t % chunks) << 2;
    const float4 v = *(const float4*)(src + (size_t)sidx[e] * F + c);
    float* d = dst + (size_t)didx[e] * F + c;
    atomicAdd(d + 0, v.x);
    atomicAdd(d + 1, v.y);
    atomicAdd(d + 2, v.z);
    atomicAdd(d + 3, v.w);
}

// x[r,:] *= (count[r] > 0 ? 1/count[r] : 0)
__global__ __launch_bounds__(256)
void k_scale_inv(float* __restrict__ x, const float* __restrict__ count, int rows, int F) {
    int i = blockIdx.x * 256 + threadIdx.x;
    if (i < rows * F) {
        float c = count[i / F];
        x[i] = (c > 0.0f) ? x[i] / c : 0.0f;
    }
}

// out = relu(out * Dinv + b)
__global__ __launch_bounds__(256)
void k_finish(float* __restrict__ out, const float* __restrict__ deg,
              const float* __restrict__ b, int rows, int F) {
    int i = blockIdx.x * 256 + threadIdx.x;
    if (i < rows * F) {
        float d = deg[i / F];
        float v = (d > 0.0f) ? out[i] / d : 0.0f;
        v += b[i % F];
        out[i] = v > 0.0f ? v : 0.0f;
    }
}

// ---------------------------------------------------------------- launch
static inline int cdiv(long a, long b) { return (int)((a + b - 1) / b); }

extern "C" void kernel_launch(void* const* d_in, const int* in_sizes, int n_in,
                              void* d_out, int out_size, void* d_ws, size_t ws_size,
                              hipStream_t stream) {
    const float* x    = (const float*)d_in[0];
    const int*   nidx = (const int*)d_in[1];            // edge[0]
    const int*   hidx = (const int*)d_in[1] + NNZ;      // edge[1]
    const float* W1   = (const float*)d_in[2];
    const float* b1   = (const float*)d_in[3];
    const float* W2   = (const float*)d_in[4];
    const float* b2   = (const float*)d_in[5];
    float* out = (float*)d_out;

    // workspace layout (256B aligned)
    const size_t OFF_DEG = 0;
    const size_t OFF_CNT = OFF_DEG + 200192;                       // 50000 f32
    const size_t OFF_ABF = OFF_CNT + 20224;                        // 5000 f32
    const size_t OFF_WT  = OFF_ABF + (size_t)N_NODES * D1 * 2;     // bf16 activations (max 50000x256)
    const size_t OFF_P   = OFF_WT  + 131072;                       // bf16 weights (transposed)
    const size_t OFF_E   = OFF_P   + (size_t)N_NODES * D1 * 4;     // f32 dense buffer (max 50000x256)
    const size_t NEED    = OFF_E   + (size_t)N_HEDGES * D1 * 4;    // f32 hedge buffer (max 5000x256)
    if (ws_size < NEED) return;

    char* ws = (char*)d_ws;
    float*  deg  = (float*)(ws + OFF_DEG);
    float*  cnt  = (float*)(ws + OFF_CNT);
    __bf16* Abf  = (__bf16*)(ws + OFF_ABF);
    __bf16* Wt   = (__bf16*)(ws + OFF_WT);
    float*  bufP = (float*)(ws + OFF_P);
    float*  ebuf = (float*)(ws + OFF_E);

    // degrees
    k_fill_zero<<<cdiv(N_NODES, 256), 256, 0, stream>>>(deg, N_NODES);
    k_fill_zero<<<cdiv(N_HEDGES, 256), 256, 0, stream>>>(cnt, N_HEDGES);
    k_degrees<<<cdiv(NNZ, 256), 256, 0, stream>>>(nidx, hidx, deg, cnt);

    // ---- layer 1: xw = x @ W1 -----------------------------------------
    k_cvt_bf16<<<cdiv((long)N_NODES * IN_C, 256), 256, 0, stream>>>(x, Abf, N_NODES * IN_C);
    k_cvt_transpose_bf16<<<cdiv(IN_C * D1, 256), 256, 0, stream>>>(W1, Wt, IN_C, D1);
    {
        int tiles = (N_NODES / 16) * (D1 / 16);
        k_gemm_bf16_wmma<<<cdiv(tiles, 8), 256, 0, stream>>>(Abf, Wt, bufP, N_NODES, D1, IN_C);
    }
    // node -> hedge
    k_fill_zero<<<cdiv((long)N_HEDGES * D1, 256), 256, 0, stream>>>(ebuf, N_HEDGES * D1);
    k_scatter_add<<<cdiv((long)NNZ * (D1 / 4), 256), 256, 0, stream>>>(bufP, nidx, ebuf, hidx, NNZ, D1);
    k_scale_inv<<<cdiv((long)N_HEDGES * D1, 256), 256, 0, stream>>>(ebuf, cnt, N_HEDGES, D1);
    // hedge -> node (reuse bufP as accumulator; xw no longer needed)
    k_fill_zero<<<cdiv((long)N_NODES * D1, 256), 256, 0, stream>>>(bufP, N_NODES * D1);
    k_scatter_add<<<cdiv((long)NNZ * (D1 / 4), 256), 256, 0, stream>>>(ebuf, hidx, bufP, nidx, NNZ, D1);
    k_finish<<<cdiv((long)N_NODES * D1, 256), 256, 0, stream>>>(bufP, deg, b1, N_NODES, D1);

    // ---- layer 2: hw = h @ W2 -----------------------------------------
    k_cvt_bf16<<<cdiv((long)N_NODES * D1, 256), 256, 0, stream>>>(bufP, Abf, N_NODES * D1);
    k_cvt_transpose_bf16<<<cdiv(D1 * D2, 256), 256, 0, stream>>>(W2, Wt, D1, D2);
    {
        int tiles = (N_NODES / 16) * (D2 / 16);
        k_gemm_bf16_wmma<<<cdiv(tiles, 8), 256, 0, stream>>>(Abf, Wt, bufP, N_NODES, D2, D1);
    }
    // node -> hedge
    k_fill_zero<<<cdiv((long)N_HEDGES * D2, 256), 256, 0, stream>>>(ebuf, N_HEDGES * D2);
    k_scatter_add<<<cdiv((long)NNZ * (D2 / 4), 256), 256, 0, stream>>>(bufP, nidx, ebuf, hidx, NNZ, D2);
    k_scale_inv<<<cdiv((long)N_HEDGES * D2, 256), 256, 0, stream>>>(ebuf, cnt, N_HEDGES, D2);
    // hedge -> node, straight into d_out
    k_fill_zero<<<cdiv((long)N_NODES * D2, 256), 256, 0, stream>>>(out, N_NODES * D2);
    k_scatter_add<<<cdiv((long)NNZ * (D2 / 4), 256), 256, 0, stream>>>(ebuf, hidx, out, nidx, NNZ, D2);
    k_finish<<<cdiv((long)N_NODES * D2, 256), 256, 0, stream>>>(out, deg, b2, N_NODES, D2);
}
